// GINNet_8340826488981
// MI455X (gfx1250) — matrix operand order repaired and checked by reading, not compiled, not run on hardware
//
#include <hip/hip_runtime.h>
#include <hip/hip_bf16.h>

typedef __attribute__((ext_vector_type(16))) _Float16 v16h;
typedef __attribute__((ext_vector_type(8)))  float    v8f;

#define HDIM 128
#define BN_EPS 1e-5f

// ---------------------------------------------------------------------------
// Zero-fill
// ---------------------------------------------------------------------------
__global__ void zero_kernel(float* __restrict__ p, size_t n) {
    size_t i = (size_t)blockIdx.x * blockDim.x + threadIdx.x;
    if (i < n) p[i] = 0.0f;
}

// ---------------------------------------------------------------------------
// Column statistics (sum, sumsq) over rows of a [nrows, 128] matrix.
// ---------------------------------------------------------------------------
__global__ void colstats_kernel(const float* __restrict__ X, int nrows,
                                int rows_per_block,
                                float* __restrict__ sum, float* __restrict__ sumsq) {
    int c  = threadIdx.x;                 // 0..127
    int r0 = blockIdx.x * rows_per_block;
    int r1 = r0 + rows_per_block; if (r1 > nrows) r1 = nrows;
    float s = 0.0f, s2 = 0.0f;
    for (int r = r0; r < r1; ++r) {
        float v = X[(size_t)r * HDIM + c];
        s += v; s2 += v * v;
    }
    atomicAdd(&sum[c], s);
    atomicAdd(&sumsq[c], s2);
}

// ---------------------------------------------------------------------------
// BN finalize: scale = gamma*rsqrt(var+eps), bias = beta - mean*scale
// ---------------------------------------------------------------------------
__global__ void bn_finalize_kernel(const float* __restrict__ sum,
                                   const float* __restrict__ sumsq,
                                   const float* __restrict__ gamma,
                                   const float* __restrict__ beta,
                                   float count,
                                   float* __restrict__ scale, float* __restrict__ bias) {
    int c = threadIdx.x;
    float m   = sum[c] / count;
    float var = sumsq[c] / count - m * m;
    float rs  = rsqrtf(var + BN_EPS);
    float sc  = gamma[c] * rs;
    scale[c] = sc;
    bias[c]  = beta[c] - m * sc;
}

// ---------------------------------------------------------------------------
// Edge scatter-add: buf[dst] += h[src]. One wave32 per edge.
// ---------------------------------------------------------------------------
__global__ void scatter_add_kernel(const float* __restrict__ h,
                                   const int* __restrict__ ei, int E,
                                   float* __restrict__ buf) {
    int gwave = (int)(((size_t)blockIdx.x * blockDim.x + threadIdx.x) >> 5);
    int lane  = threadIdx.x & 31;
    if (gwave >= E) return;
    int src = ei[gwave];
    int dst = ei[E + gwave];
    const float4 v = ((const float4*)(h + (size_t)src * HDIM))[lane];
    float* bp = buf + (size_t)dst * HDIM + lane * 4;
    atomicAdd(bp + 0, v.x); atomicAdd(bp + 1, v.y);
    atomicAdd(bp + 2, v.z); atomicAdd(bp + 3, v.w);
}

// ---------------------------------------------------------------------------
// Graph pooling: g[batch[n]] += h[n]. One wave32 per node.
// ---------------------------------------------------------------------------
__global__ void pool_kernel(const float* __restrict__ h,
                            const int* __restrict__ batch, int N,
                            float* __restrict__ g) {
    int gwave = (int)(((size_t)blockIdx.x * blockDim.x + threadIdx.x) >> 5);
    int lane  = threadIdx.x & 31;
    if (gwave >= N) return;
    int seg = batch[gwave];
    const float4 v = ((const float4*)(h + (size_t)gwave * HDIM))[lane];
    float* gp = g + (size_t)seg * HDIM + lane * 4;
    atomicAdd(gp + 0, v.x); atomicAdd(gp + 1, v.y);
    atomicAdd(gp + 2, v.z); atomicAdd(gp + 3, v.w);
}

// ---------------------------------------------------------------------------
// WMMA GEMM, fully specialized by template flags:
//   Out[r,n] = f( map(A[r,:] (+A2[r,:])) @ W + out_bias[n] ), K = Ncols = 128
//   HAS_MAP:  a = a*in_scale[k] + in_bias[k]      (BatchNorm on input)
//   INRELU:   a = max(a,0) after map
//   OUTRELU:  o = max(o,0)
//   STATS:    accumulate per-column sum / sumsq of stored output (valid rows)
// Block: 256 threads = 8 wave32; covers 128 rows x 128 cols of output.
// W (f32, row-major [k][n]) is transposed into f16 LDS once per block.
// ---------------------------------------------------------------------------
template <bool HAS_A2, bool HAS_MAP, bool INRELU, bool OUTRELU, bool STATS>
__global__ __launch_bounds__(256)
void gemm128_wmma_kernel(const float* __restrict__ A,
                         const float* __restrict__ A2, int nrows,
                         const float* __restrict__ W,
                         const float* __restrict__ in_scale,
                         const float* __restrict__ in_bias,
                         const float* __restrict__ out_bias,
                         float* __restrict__ Out,
                         float* __restrict__ stat_sum, float* __restrict__ stat_sq) {
    __shared__ _Float16 ldsW[HDIM * HDIM];   // transposed: ldsW[n*128 + k], 32 KB

    int tid = threadIdx.x;
    // W[k][n] (f32) -> ldsW[n][k] (f16)
    for (int i = tid; i < HDIM * HDIM; i += 256) {
        int k = i >> 7, n = i & (HDIM - 1);
        ldsW[n * HDIM + k] = (_Float16)W[i];
    }
    __syncthreads();

    const int wave  = tid >> 5;
    const int lane  = tid & 31;
    const int lrow  = lane & 15;       // row-in-tile (A), col-in-tile (B/C)
    const int lhalf = lane >> 4;       // lane half selects K/M sub-ranges
    const int rowbase = blockIdx.x * 128 + wave * 16;
    // Block-uniform (scalar) fast-path predicate: last block is the only tail.
    const bool fullblock = ((int)blockIdx.x * 128 + 128) <= nrows;
    int row  = rowbase + lrow;
    int rowc = row < nrows ? row : (nrows - 1);      // clamp loads; stores masked

    // ---- A fragments: 4 chunks of 16x32 f16 (ISA 7.12.2 layout) ----
    // lanes 0-15 : row M=lrow, halves 0..7 = K base+0..7,  8..15 = K base+16..23
    // lanes 16-31: row M=lrow, halves 0..7 = K base+8..15, 8..15 = K base+24..31
    v16h afrag[4];
    const float* Arow  = A + (size_t)rowc * HDIM;
    const float* A2row = HAS_A2 ? A2 + (size_t)rowc * HDIM : (const float*)0;
#pragma unroll
    for (int kk = 0; kk < 4; ++kk) {
        int k0 = kk * 32 + lhalf * 8;
#pragma unroll
        for (int s = 0; s < 2; ++s) {
            int kb = k0 + s * 16;    // multiple of 8 floats -> 16B aligned
            float4 v0 = *(const float4*)(Arow + kb);
            float4 v1 = *(const float4*)(Arow + kb + 4);
            float a[8] = {v0.x, v0.y, v0.z, v0.w, v1.x, v1.y, v1.z, v1.w};
            if (HAS_A2) {
                float4 w0 = *(const float4*)(A2row + kb);
                float4 w1 = *(const float4*)(A2row + kb + 4);
                a[0] += w0.x; a[1] += w0.y; a[2] += w0.z; a[3] += w0.w;
                a[4] += w1.x; a[5] += w1.y; a[6] += w1.z; a[7] += w1.w;
            }
#pragma unroll
            for (int j = 0; j < 8; ++j) {
                float av = a[j];
                if (HAS_MAP) av = av * in_scale[kb + j] + in_bias[kb + j];
                if (INRELU)  av = fmaxf(av, 0.0f);
                afrag[kk][s * 8 + j] = (_Float16)av;
            }
        }
    }

    // ---- preload this lane's 8 output-bias values (one per tile) ----
    float bn[8];
#pragma unroll
    for (int nt = 0; nt < 8; ++nt) bn[nt] = out_bias[nt * 16 + lrow];

    // ---- 8 output tiles of 16x16 across the 128 columns ----
#pragma unroll 1
    for (int nt = 0; nt < 8; ++nt) {
        int ncol = nt * 16 + lrow;     // this lane's output column
        // Load all four B fragments first (one LDS clause), then WMMA burst.
        // B 32x16 f16: lanes 0-15 K=0..7,16..23, lanes 16-31 K=8..15,24..31, col=lane%16
        v16h bfrag[4];
#pragma unroll
        for (int kk = 0; kk < 4; ++kk) {
            const _Float16* wp = &ldsW[ncol * HDIM + kk * 32 + lhalf * 8];
#pragma unroll
            for (int j = 0; j < 16; ++j) bfrag[kk][j] = wp[j < 8 ? j : j + 8];
        }
        v8f acc = {};
#pragma unroll
        for (int kk = 0; kk < 4; ++kk) {
            acc = __builtin_amdgcn_wmma_f32_16x16x32_f16(
                false, afrag[kk], false, bfrag[kk], (short)0, acc, false, false);
        }
        // C/D layout: lane col = lane%16; VGPR i -> row i (lanes<16), 8+i (>=16)
        float s1 = 0.0f, s2 = 0.0f;
        float* outcol = Out + (size_t)(rowbase + lhalf * 8) * HDIM + ncol;
        if (fullblock) {
#pragma unroll
            for (int i = 0; i < 8; ++i) {
                float v = acc[i] + bn[nt];
                if (OUTRELU) v = fmaxf(v, 0.0f);
                outcol[(size_t)i * HDIM] = v;
                if (STATS) { s1 += v; s2 += v * v; }
            }
        } else {
#pragma unroll
            for (int i = 0; i < 8; ++i) {
                int r = rowbase + lhalf * 8 + i;
                float v = acc[i] + bn[nt];
                if (OUTRELU) v = fmaxf(v, 0.0f);
                if (r < nrows) {
                    outcol[(size_t)i * HDIM] = v;
                    if (STATS) { s1 += v; s2 += v * v; }
                }
            }
        }
        if (STATS) {
            atomicAdd(&stat_sum[ncol], s1);
            atomicAdd(&stat_sq[ncol], s2);
        }
    }
}

// ---------------------------------------------------------------------------
// Head: logits = bn(g2) @ W_cls + b_cls, then log_softmax. One thread / graph.
// ---------------------------------------------------------------------------
__global__ void head_kernel(const float* __restrict__ g2,
                            const float* __restrict__ scale,
                            const float* __restrict__ bias,
                            const float* __restrict__ Wc,
                            const float* __restrict__ bc,
                            int nG, int C, float* __restrict__ out) {
    int row = blockIdx.x * blockDim.x + threadIdx.x;
    if (row >= nG) return;
    float logit[16];
    for (int c = 0; c < C; ++c) logit[c] = bc[c];
    for (int k = 0; k < HDIM; ++k) {
        float v = g2[(size_t)row * HDIM + k] * scale[k] + bias[k];
        for (int c = 0; c < C; ++c) logit[c] += v * Wc[k * C + c];
    }
    float mx = logit[0];
    for (int c = 1; c < C; ++c) mx = fmaxf(mx, logit[c]);
    float se = 0.0f;
    for (int c = 0; c < C; ++c) se += expf(logit[c] - mx);
    float lse = mx + logf(se);
    for (int c = 0; c < C; ++c) out[(size_t)row * C + c] = logit[c] - lse;
}

// ---------------------------------------------------------------------------
// Launcher
// ---------------------------------------------------------------------------
extern "C" void kernel_launch(void* const* d_in, const int* in_sizes, int n_in,
                              void* d_out, int out_size, void* d_ws, size_t ws_size,
                              hipStream_t stream) {
    // setup_inputs() dict order
    const float* x          = (const float*)d_in[0];
    const int*   ei         = (const int*)  d_in[1];
    const int*   batch      = (const int*)  d_in[2];
    const float* gamma_feat = (const float*)d_in[3];
    const float* beta_feat  = (const float*)d_in[4];
    const float* W_feat     = (const float*)d_in[5];
    const float* b_feat     = (const float*)d_in[6];
    const float* gin_W1     = (const float*)d_in[7];
    const float* gin_b1     = (const float*)d_in[8];
    const float* gin_gamma  = (const float*)d_in[9];
    const float* gin_beta   = (const float*)d_in[10];
    const float* gin_W2     = (const float*)d_in[11];
    const float* gin_b2     = (const float*)d_in[12];
    const float* gamma_fc   = (const float*)d_in[13];
    const float* beta_fc    = (const float*)d_in[14];
    const float* W_fc       = (const float*)d_in[15];
    const float* b_fc       = (const float*)d_in[16];
    const float* gamma_h    = (const float*)d_in[17];
    const float* beta_h     = (const float*)d_in[18];
    const float* W_cls      = (const float*)d_in[19];
    const float* b_cls      = (const float*)d_in[20];

    const int N  = in_sizes[0] / HDIM;      // 50000
    const int E  = in_sizes[1] / 2;         // 600000
    const int C  = in_sizes[20];            // 10
    const int nG = out_size / C;            // 500
    const int NUM_GIN = in_sizes[8] / HDIM; // 3

    // workspace carve-up (floats)
    float* ws    = (float*)d_ws;
    float* hbuf  = ws;                                  // [N,128]
    float* buf   = hbuf  + (size_t)N * HDIM;            // [N,128] (aggregation)
    float* z1    = buf   + (size_t)N * HDIM;            // [N,128] (pre-BN)
    float* g     = z1    + (size_t)N * HDIM;            // [G,128]
    float* g2    = g     + (size_t)nG * HDIM;           // [G,128]
    float* ssum  = g2    + (size_t)nG * HDIM;           // [128]
    float* ssq   = ssum  + HDIM;                        // [128]
    float* scl   = ssq   + HDIM;                        // [128]
    float* bia   = scl   + HDIM;                        // [128]

    const int gemmBlocksN = (N + 127) / 128;
    const int gemmBlocksG = (nG + 127) / 128;
    const int statBlocksN = (N + 255) / 256;
    const int statBlocksG = (nG + 255) / 256;

    // ---- feat stage: h = relu(bn(x) @ W_feat + b_feat) ----
    zero_kernel<<<1, 256, 0, stream>>>(ssum, 2 * HDIM);
    colstats_kernel<<<statBlocksN, HDIM, 0, stream>>>(x, N, 256, ssum, ssq);
    bn_finalize_kernel<<<1, HDIM, 0, stream>>>(ssum, ssq, gamma_feat, beta_feat,
                                               (float)N, scl, bia);
    gemm128_wmma_kernel<false, true, false, true, false>
        <<<gemmBlocksN, 256, 0, stream>>>(
        x, (const float*)0, N, W_feat, scl, bia, b_feat,
        hbuf, (float*)0, (float*)0);

    // ---- GIN layers ----
    for (int i = 0; i < NUM_GIN; ++i) {
        const float* W1 = gin_W1 + (size_t)i * HDIM * HDIM;
        const float* B1 = gin_b1 + (size_t)i * HDIM;
        const float* Gm = gin_gamma + (size_t)i * HDIM;
        const float* Bt = gin_beta  + (size_t)i * HDIM;
        const float* W2 = gin_W2 + (size_t)i * HDIM * HDIM;
        const float* B2 = gin_b2 + (size_t)i * HDIM;

        // agg = scatter_add(h[src] -> dst);  z_in = h + agg (fused in GEMM A2)
        zero_kernel<<<(int)(((size_t)N * HDIM + 255) / 256), 256, 0, stream>>>(
            buf, (size_t)N * HDIM);
        scatter_add_kernel<<<(E + 7) / 8, 256, 0, stream>>>(hbuf, ei, E, buf);

        // z1 = (h + agg) @ W1 + b1, fused column stats for BN
        zero_kernel<<<1, 256, 0, stream>>>(ssum, 2 * HDIM);
        gemm128_wmma_kernel<true, false, false, false, true>
            <<<gemmBlocksN, 256, 0, stream>>>(
            hbuf, buf, N, W1, (const float*)0, (const float*)0, B1,
            z1, ssum, ssq);
        bn_finalize_kernel<<<1, HDIM, 0, stream>>>(ssum, ssq, Gm, Bt,
                                                   (float)N, scl, bia);

        // h = relu( relu(bn(z1)) @ W2 + b2 )
        gemm128_wmma_kernel<false, true, true, true, false>
            <<<gemmBlocksN, 256, 0, stream>>>(
            z1, (const float*)0, N, W2, scl, bia, B2,
            hbuf, (float*)0, (float*)0);
    }

    // ---- pooling: g[batch[n]] += h[n] ----
    zero_kernel<<<(int)(((size_t)nG * HDIM + 255) / 256), 256, 0, stream>>>(
        g, (size_t)nG * HDIM);
    pool_kernel<<<(N + 7) / 8, 256, 0, stream>>>(hbuf, batch, N, g);

    // ---- fc: g2 = relu(bn(g) @ W_fc + b_fc), fused stats of g2 ----
    zero_kernel<<<1, 256, 0, stream>>>(ssum, 2 * HDIM);
    colstats_kernel<<<statBlocksG, HDIM, 0, stream>>>(g, nG, 256, ssum, ssq);
    bn_finalize_kernel<<<1, HDIM, 0, stream>>>(ssum, ssq, gamma_fc, beta_fc,
                                               (float)nG, scl, bia);
    zero_kernel<<<1, 256, 0, stream>>>(ssum, 2 * HDIM);
    gemm128_wmma_kernel<false, true, false, true, true>
        <<<gemmBlocksG, 256, 0, stream>>>(
        g, (const float*)0, nG, W_fc, scl, bia, b_fc,
        g2, ssum, ssq);
    bn_finalize_kernel<<<1, HDIM, 0, stream>>>(ssum, ssq, gamma_h, beta_h,
                                               (float)nG, scl, bia);

    // ---- classifier + log_softmax ----
    head_kernel<<<(nG + 63) / 64, 64, 0, stream>>>(g2, scl, bia, W_cls, b_cls,
                                                   nG, C, (float*)d_out);
}
